// LstmHandSynthesis_74199855005990
// MI455X (gfx1250) — compile-verified, matrix operand order checked by reading, not run on hardware
//
#include <hip/hip_runtime.h>

#define Bsz  64
#define Tlen 1191
#define Hdim 400
#define G4   1600          // 4*H gates
#define K1   480           // layer-1 A width: 400(h)+3(x)+72(wind)+5 pad
#define KH   416           // padded H for recurrence / window K
#define KF   1216          // padded 3H for FC
#define NWG  25            // persistent workgroups (one per 16-wide H slice)

typedef __bf16 bf16;
typedef __attribute__((ext_vector_type(16))) __bf16 v16bf;
typedef __attribute__((ext_vector_type(8)))  __bf16 v8bf;
typedef __attribute__((ext_vector_type(8)))  float  v8f;

union FragU { v16bf v; v8bf h[2]; };

// A/B fragment gather (ISA 7.12.2 16-bit layout): lane L -> row = tile*16 + L%16,
// K values = kc + (L>=16 ? 8:0) + {0..7}  and  kc + 16 + (L>=16 ? 8:0) + {0..7}
__device__ __forceinline__ v16bf load_frag(const bf16* __restrict__ base, int row,
                                           int stride, int kc, int lane) {
  const bf16* p = base + (size_t)row * stride + kc + ((lane & 16) ? 8 : 0);
  FragU f;
  f.h[0] = *(const v8bf*)p;
  f.h[1] = *(const v8bf*)(p + 16);
  return f.v;
}

__device__ __forceinline__ v8f wmma_bf(v16bf a, v16bf b, v8f c) {
  return __builtin_amdgcn_wmma_f32_16x16x32_bf16(false, a, false, b, (short)0, c,
                                                 false, false);
}

__device__ __forceinline__ v8f splat8(float x) {
  v8f r;
#pragma unroll
  for (int i = 0; i < 8; ++i) r[i] = x;
  return r;
}

__device__ __forceinline__ float sigm(float x) { return 1.f / (1.f + __expf(-x)); }

// Optimization barrier: keeps loop-invariant weight loads *inside* the t-loop
// (prevents LICM from hoisting ~400 VGPRs of fragments and spilling to scratch).
__device__ __forceinline__ int opaque_i(int x) {
  asm volatile("" : "+v"(x));
  return x;
}

// Device-scope grid barrier for the persistent RNN kernels.
__device__ __forceinline__ void grid_sync(int* cnt, int expect) {
  __syncthreads();
  if (threadIdx.x == 0) {
    __threadfence();
    __hip_atomic_fetch_add(cnt, 1, __ATOMIC_ACQ_REL, __HIP_MEMORY_SCOPE_AGENT);
    while (__hip_atomic_load(cnt, __ATOMIC_ACQUIRE, __HIP_MEMORY_SCOPE_AGENT) < expect)
      __builtin_amdgcn_s_sleep(2);
  }
  __syncthreads();
}

// ---------------- weight packing (f32 -> padded bf16, row-major) ----------------
__global__ void pad_convert(const float* __restrict__ src, bf16* __restrict__ dst,
                            int srows, int scols, int drows, int dcols) {
  int n = drows * dcols;
  for (int i = blockIdx.x * blockDim.x + threadIdx.x; i < n;
       i += gridDim.x * blockDim.x) {
    int r = i / dcols, c = i % dcols;
    float v = (r < srows && c < scols) ? src[(size_t)r * scols + c] : 0.f;
    dst[i] = (bf16)v;
  }
}

// layer1 combined weight: [Whh1(400) | Wih1_x(3) | Wih1_wind(72) | pad]; bias1 = bih1+bhh1
__global__ void pack_wcomb1(const float* __restrict__ Wih1, const float* __restrict__ Whh1,
                            const float* __restrict__ bih1, const float* __restrict__ bhh1,
                            bf16* __restrict__ Wc, float* __restrict__ bias1) {
  int n = G4 * K1;
  for (int i = blockIdx.x * blockDim.x + threadIdx.x; i < n;
       i += gridDim.x * blockDim.x) {
    int r = i / K1, k = i % K1;
    float v = (k < Hdim) ? Whh1[(size_t)r * Hdim + k]
                         : ((k < 475) ? Wih1[(size_t)r * 75 + (k - Hdim)] : 0.f);
    Wc[i] = (bf16)v;
    if (k == 0) bias1[r] = bih1[r] + bhh1[r];
  }
}

// ---------------- init: zero h-images / counters, seed x0, one-hot indices ----------------
__global__ void init_kernel(const float* __restrict__ stroke, const float* __restrict__ hot,
                            bf16* __restrict__ hbuf1, bf16* __restrict__ hb2,
                            bf16* __restrict__ hb3, int* __restrict__ cidx,
                            int* __restrict__ syncs) {
  int idx0 = blockIdx.x * blockDim.x + threadIdx.x;
  int stride = gridDim.x * blockDim.x;
  for (int i = idx0; i < 2 * 64 * K1; i += stride) {
    int buf = i / (64 * K1), r = i % (64 * K1);
    int b = r / K1, k = r % K1;
    float v = 0.f;
    if (buf == 0 && k >= Hdim && k < Hdim + 3)
      v = stroke[((size_t)b * Tlen) * 3 + (k - Hdim)];
    hbuf1[i] = (bf16)v;
  }
  for (int i = idx0; i < 2 * 64 * KH; i += stride) {
    hb2[i] = (bf16)0.f;
    hb3[i] = (bf16)0.f;
  }
  for (int i = idx0; i < 64 * 64; i += stride) {   // argmax of one-hot rows
    const float* row = hot + (size_t)i * 72;
    int c = 0;
    for (int k = 0; k < 72; ++k)
      if (row[k] > 0.5f) c = k;
    cidx[i] = c;
  }
  if (idx0 < 4) syncs[idx0] = 0;
}

// ---------------- layer 1: multi-WG persistent LSTM + attention ----------------
// grid = 25 WGs x 128 threads (4 waves); WG jt owns h columns [jt*16, jt*16+16).
// Weights live in LDS for all steps; h image staged LDS-side each step.
__global__ __launch_bounds__(128, 1) void layer1_persist(
    const float* __restrict__ stroke,
    const bf16* __restrict__ Wc, const float* __restrict__ bias1,
    const bf16* __restrict__ Wwinp, const float* __restrict__ bwin,
    const int* __restrict__ cidx,
    bf16* __restrict__ hbuf, bf16* __restrict__ h1_seq, bf16* __restrict__ wind_seq,
    int* __restrict__ cnt) {
  extern __shared__ char smem[];
  bf16*  Wl  = (bf16*)smem;                              // 64 x 480 gate weights
  bf16*  Al  = (bf16*)(smem + 64 * K1 * 2);              // 64 x 480 A image
  bf16*  Ww  = (bf16*)(smem + 2 * 64 * K1 * 2);          // 32 x 416 window weights (WG0)
  float* lin = (float*)(smem + 2 * 64 * K1 * 2 + 32 * KH * 2);
  float* phi = lin + 64 * 32;
  float* kap = phi + 64 * 64;
  float* aal = kap + 64 * 16;
  float* bbe = aal + 64 * 16;
  float* kne = bbe + 64 * 16;
  float* wnd = kne + 64 * 16;                            // 64 x 72

  const int tid = threadIdx.x, lane = tid & 31, wave = tid >> 5;
  const int nloc = lane & 15, hi8 = (lane & 16) ? 8 : 0;
  const int jt = blockIdx.x;                             // 0..24
  const int m = wave;                                    // m-tile (batch slice)

  // stage this WG's gate-weight slice: Wl[g*16+r][:] = Wc[g*400 + jt*16 + r][:]
  for (int i = tid; i < 64 * (K1 / 8); i += 128) {
    int r = i / (K1 / 8), c8 = i % (K1 / 8);
    int g = r >> 4, rr = r & 15;
    ((v8bf*)Wl)[i] =
        *(const v8bf*)(Wc + ((size_t)(g * Hdim + jt * 16 + rr)) * K1 + c8 * 8);
  }
  if (jt == 0) {
    for (int i = tid; i < 32 * (KH / 8); i += 128) {
      int r = i / (KH / 8), c8 = i % (KH / 8);
      ((v8bf*)Ww)[i] = *(const v8bf*)(Wwinp + (size_t)r * KH + c8 * 8);
    }
    for (int i = tid; i < 64 * 16; i += 128) kap[i] = 0.f;
  }
  __syncthreads();

  float bsc[4];
#pragma unroll
  for (int g = 0; g < 4; ++g) bsc[g] = bias1[g * Hdim + jt * 16 + nloc];

  v8f cst = splat8(0.f);

  for (int t = 0; t < Tlen; ++t) {
    const bf16* gcur = hbuf + (t & 1) * (64 * K1);
    bf16*       gnxt = hbuf + ((t + 1) & 1) * (64 * K1);

    // stage full A image [h | x | wind | pad] into LDS
    for (int i = tid; i < 64 * (K1 / 8); i += 128)
      ((v8bf*)Al)[i] = ((const v8bf*)gcur)[i];
    __syncthreads();

    const int zz = opaque_i(0);   // pin weight loads in-loop
    v8f a0 = splat8(bsc[0]), a1 = splat8(bsc[1]);
    v8f a2 = splat8(bsc[2]), a3 = splat8(bsc[3]);
#pragma unroll
    for (int kc = 0; kc < K1; kc += 32) {
      v16bf af = load_frag(Al, m * 16 + nloc, K1, kc, lane);
      a0 = wmma_bf(af, load_frag(Wl,  0 + nloc + zz, K1, kc, lane), a0);
      a1 = wmma_bf(af, load_frag(Wl, 16 + nloc + zz, K1, kc, lane), a1);
      a2 = wmma_bf(af, load_frag(Wl, 32 + nloc + zz, K1, kc, lane), a2);
      a3 = wmma_bf(af, load_frag(Wl, 48 + nloc + zz, K1, kc, lane), a3);
    }
#pragma unroll
    for (int e = 0; e < 8; ++e) {
      float ig = sigm(a0[e]), fg = sigm(a1[e]);
      float gg = tanhf(a2[e]), og = sigm(a3[e]);
      float cc = fg * cst[e] + ig * gg;
      cst[e] = cc;
      float hh = og * tanhf(cc);
      int b = m * 16 + e + hi8, j = jt * 16 + nloc;
      gnxt[b * K1 + j] = (bf16)hh;
      h1_seq[((size_t)t * Bsz + b) * Hdim + j] = (bf16)hh;
    }
    grid_sync(cnt, NWG * (2 * t + 1));

    if (jt == 0) {
      // lin = h1_t @ Wwin^T + bwin  (wave = m-tile, 2 n-tiles each)
#pragma unroll
      for (int ntw = 0; ntw < 2; ++ntw) {
        int col = ntw * 16 + nloc;
        v8f acc = splat8((col < 30) ? bwin[col] : 0.f);
#pragma unroll
        for (int kc = 0; kc < KH; kc += 32) {
          v16bf af  = load_frag(gnxt, m * 16 + nloc, K1, kc, lane);
          v16bf bfr = load_frag(Ww, col + zz, KH, kc, lane);
          acc = wmma_bf(af, bfr, acc);
        }
#pragma unroll
        for (int e = 0; e < 8; ++e) lin[(m * 16 + e + hi8) * 32 + col] = acc[e];
      }
      for (int i = tid; i < 64 * 72; i += 128) wnd[i] = 0.f;
      __syncthreads();
      for (int i = tid; i < 640; i += 128) {
        int b = i / 10, k = i % 10;
        aal[b * 16 + k] = __expf(lin[b * 32 + k]);
        bbe[b * 16 + k] = __expf(lin[b * 32 + 10 + k]);
        kne[b * 16 + k] = kap[b * 16 + k] + 0.05f * __expf(lin[b * 32 + 20 + k]);
      }
      __syncthreads();
      for (int i = tid; i < 64 * 64; i += 128) {
        int b = i >> 6, u = i & 63;
        float s = 0.f;
#pragma unroll
        for (int k = 0; k < 10; ++k) {
          float d = kne[b * 16 + k] - (float)u;
          s += aal[b * 16 + k] * __expf(-bbe[b * 16 + k] * d * d);
        }
        phi[i] = s;
      }
      __syncthreads();
      for (int i = tid; i < 640; i += 128) {
        int b = i / 10, k = i % 10;
        kap[b * 16 + k] = kne[b * 16 + k];
      }
      // wind = phi @ one_hot  == scatter-add of phi into char columns (exact)
      for (int i = tid; i < 64 * 64; i += 128)
        atomicAdd(&wnd[(i >> 6) * 72 + cidx[i]], phi[i]);
      __syncthreads();
      for (int i = tid; i < 64 * 72; i += 128) {
        int b = i / 72, c = i % 72;
        float s = wnd[i];
        gnxt[b * K1 + 403 + c] = (bf16)s;
        wind_seq[((size_t)t * Bsz + b) * 72 + c] = (bf16)s;
      }
      if (t + 1 < Tlen) {
        for (int i = tid; i < 192; i += 128) {
          int b = i / 3, cmp = i % 3;
          gnxt[b * K1 + Hdim + cmp] =
              (bf16)stroke[((size_t)b * Tlen + t + 1) * 3 + cmp];
        }
      }
    }
    grid_sync(cnt, NWG * (2 * t + 2));
  }
}

// ---------------- layers 2/3: multi-WG persistent recurrence ----------------
__global__ __launch_bounds__(128, 1) void rec_persist(
    const float* __restrict__ G, const bf16* __restrict__ Whhp,
    bf16* __restrict__ hbuf, bf16* __restrict__ h_seq, int* __restrict__ cnt) {
  extern __shared__ char smem[];
  bf16* Wl = (bf16*)smem;                    // 64 x 416 gate weights (resident)
  bf16* Al = (bf16*)(smem + 64 * KH * 2);    // 64 x 416 A image

  const int tid = threadIdx.x, lane = tid & 31, wave = tid >> 5;
  const int nloc = lane & 15, hi8 = (lane & 16) ? 8 : 0;
  const int jt = blockIdx.x;
  const int m = wave;

  for (int i = tid; i < 64 * (KH / 8); i += 128) {
    int r = i / (KH / 8), c8 = i % (KH / 8);
    int g = r >> 4, rr = r & 15;
    ((v8bf*)Wl)[i] =
        *(const v8bf*)(Whhp + ((size_t)(g * Hdim + jt * 16 + rr)) * KH + c8 * 8);
  }
  __syncthreads();

  v8f cst = splat8(0.f);

  for (int t = 0; t < Tlen; ++t) {
    const bf16* gcur = hbuf + (t & 1) * (64 * KH);
    bf16*       gnxt = hbuf + ((t + 1) & 1) * (64 * KH);
    for (int i = tid; i < 64 * (KH / 8); i += 128)
      ((v8bf*)Al)[i] = ((const v8bf*)gcur)[i];
    __syncthreads();

    const int zz = opaque_i(0);
    v8f a0 = splat8(0.f), a1 = splat8(0.f), a2 = splat8(0.f), a3 = splat8(0.f);
#pragma unroll
    for (int e = 0; e < 8; ++e) {   // accumulators start at precomputed pregates
      size_t rb = ((size_t)t * Bsz + m * 16 + e + hi8) * G4 + jt * 16 + nloc;
      a0[e] = G[rb];
      a1[e] = G[rb + Hdim];
      a2[e] = G[rb + 2 * Hdim];
      a3[e] = G[rb + 3 * Hdim];
    }
#pragma unroll
    for (int kc = 0; kc < KH; kc += 32) {
      v16bf af = load_frag(Al, m * 16 + nloc, KH, kc, lane);
      a0 = wmma_bf(af, load_frag(Wl,  0 + nloc + zz, KH, kc, lane), a0);
      a1 = wmma_bf(af, load_frag(Wl, 16 + nloc + zz, KH, kc, lane), a1);
      a2 = wmma_bf(af, load_frag(Wl, 32 + nloc + zz, KH, kc, lane), a2);
      a3 = wmma_bf(af, load_frag(Wl, 48 + nloc + zz, KH, kc, lane), a3);
    }
#pragma unroll
    for (int e = 0; e < 8; ++e) {
      float ig = sigm(a0[e]), fg = sigm(a1[e]);
      float gg = tanhf(a2[e]), og = sigm(a3[e]);
      float cc = fg * cst[e] + ig * gg;
      cst[e] = cc;
      float hh = og * tanhf(cc);
      int b = m * 16 + e + hi8, j = jt * 16 + nloc;
      gnxt[b * KH + j] = (bf16)hh;
      h_seq[((size_t)t * Bsz + b) * Hdim + j] = (bf16)hh;
    }
    grid_sync(cnt, NWG * (t + 1));
  }
}

// ---------------- pregate GEMM: G = [x|h|wind] @ Wih^T + (bih+bhh) ----------------
__global__ __launch_bounds__(256, 2) void pregate_gemm(
    const float* __restrict__ stroke, const bf16* __restrict__ h_seq,
    const bf16* __restrict__ wind_seq, const bf16* __restrict__ Wp,
    const float* __restrict__ bih, const float* __restrict__ bhh,
    float* __restrict__ G) {
  extern __shared__ char smem[];
  bf16* As = (bf16*)smem;                  // 16 x K1
  const int mt = blockIdx.x;
  const int tid = threadIdx.x;
  for (int i = tid; i < 16 * K1; i += 256) {
    int m = i / K1, k = i % K1;
    int row = mt * 16 + m;
    int t = row / Bsz, b = row % Bsz;
    bf16 v;
    if (k < 3)        v = (bf16)stroke[((size_t)b * Tlen + t) * 3 + k];
    else if (k < 403) v = h_seq[(size_t)row * Hdim + (k - 3)];
    else if (k < 475) v = wind_seq[(size_t)row * 72 + (k - 403)];
    else              v = (bf16)0.f;
    As[i] = v;
  }
  __syncthreads();
  const int lane = tid & 31, wave = tid >> 5, nloc = lane & 15;
  const int hi8 = (lane & 16) ? 8 : 0;
  for (int nt = wave; nt < 100; nt += 8) {
    int n = nt * 16 + nloc;
    v8f acc = splat8(bih[n] + bhh[n]);
#pragma unroll
    for (int kc = 0; kc < K1; kc += 32) {
      v16bf af  = load_frag(As, nloc, K1, kc, lane);
      v16bf bfr = load_frag(Wp, n, K1, kc, lane);
      acc = wmma_bf(af, bfr, acc);
    }
#pragma unroll
    for (int e = 0; e < 8; ++e)
      G[((size_t)mt * 16 + e + hi8) * G4 + n] = acc[e];
  }
}

// ---------------- FC GEMM: fcraw = [h1|h2|h3] @ Wfc^T + bfc ----------------
__global__ __launch_bounds__(256, 2) void fc_gemm(
    const bf16* __restrict__ h1, const bf16* __restrict__ h2,
    const bf16* __restrict__ h3, const bf16* __restrict__ Wfcp,
    const float* __restrict__ bfc, float* __restrict__ fcraw) {
  extern __shared__ char smem[];
  bf16* As = (bf16*)smem;                  // 16 x KF
  const int mt = blockIdx.x;
  const int tid = threadIdx.x;
  for (int i = tid; i < 16 * KF; i += 256) {
    int m = i / KF, k = i % KF;
    size_t row = (size_t)mt * 16 + m;
    bf16 v;
    if (k < 400)       v = h1[row * Hdim + k];
    else if (k < 800)  v = h2[row * Hdim + (k - 400)];
    else if (k < 1200) v = h3[row * Hdim + (k - 800)];
    else               v = (bf16)0.f;
    As[i] = v;
  }
  __syncthreads();
  const int lane = tid & 31, wave = tid >> 5, nloc = lane & 15;
  const int hi8 = (lane & 16) ? 8 : 0;
  int n = wave * 16 + nloc;                // 8 waves -> 8 N-tiles (128 cols)
  v8f acc = splat8((n < 121) ? bfc[n] : 0.f);
  for (int kc = 0; kc < KF; kc += 32) {
    v16bf af  = load_frag(As, nloc, KF, kc, lane);
    v16bf bfr = load_frag(Wfcp, n, KF, kc, lane);
    acc = wmma_bf(af, bfr, acc);
  }
#pragma unroll
  for (int e = 0; e < 8; ++e)
    fcraw[((size_t)mt * 16 + e + hi8) * 128 + n] = acc[e];
}

// ---------------- MDN epilogue ----------------
__global__ void out_transform(const float* __restrict__ fcraw, float* __restrict__ out) {
  int row = blockIdx.x * blockDim.x + threadIdx.x;
  if (row >= Tlen * Bsz) return;
  int t = row / Bsz, b = row % Bsz;
  const float* f = fcraw + (size_t)row * 128;
  const size_t TBn = (size_t)Tlen * Bsz;
  size_t base = (size_t)b * Tlen + t;
  out[base] = 1.f / (1.f + __expf(f[0]));               // sigmoid(-end)
  float mx = f[1];
  for (int k = 1; k < 20; ++k) mx = fmaxf(mx, f[1 + k]);
  float ev[20]; float s = 0.f;
  for (int k = 0; k < 20; ++k) { ev[k] = __expf(f[1 + k] - mx); s += ev[k]; }
  float inv = 1.f / s;
  for (int k = 0; k < 20; ++k) out[TBn + base * 20 + k] = ev[k] * inv;            // pi
  for (int k = 0; k < 40; ++k) out[21 * TBn + base * 40 + k] = f[21 + k];         // mu
  for (int k = 0; k < 40; ++k) out[61 * TBn + base * 40 + k] = __expf(f[61 + k]); // sigma
  for (int k = 0; k < 20; ++k) out[101 * TBn + base * 20 + k] = tanhf(f[101 + k]);// rho
}

// ---------------- launch ----------------
extern "C" void kernel_launch(void* const* d_in, const int* in_sizes, int n_in,
                              void* d_out, int out_size, void* d_ws, size_t ws_size,
                              hipStream_t stream) {
  (void)in_sizes; (void)n_in; (void)out_size; (void)ws_size;
  const float* stroke = (const float*)d_in[0];
  const float* hot    = (const float*)d_in[1];
  const float* W_ih1  = (const float*)d_in[2];
  const float* W_hh1  = (const float*)d_in[3];
  const float* b_ih1  = (const float*)d_in[4];
  const float* b_hh1  = (const float*)d_in[5];
  const float* W_win  = (const float*)d_in[6];
  const float* b_win  = (const float*)d_in[7];
  const float* W_ih2  = (const float*)d_in[8];
  const float* W_hh2  = (const float*)d_in[9];
  const float* b_ih2  = (const float*)d_in[10];
  const float* b_hh2  = (const float*)d_in[11];
  const float* W_ih3  = (const float*)d_in[12];
  const float* W_hh3  = (const float*)d_in[13];
  const float* b_ih3  = (const float*)d_in[14];
  const float* b_hh3  = (const float*)d_in[15];
  const float* W_fc   = (const float*)d_in[16];
  const float* b_fc   = (const float*)d_in[17];

  const size_t TB = (size_t)Tlen * Bsz;      // 76224 (multiple of 16)
  char* ws = (char*)d_ws;
  auto AL = [](size_t x) { return (x + 255) & ~(size_t)255; };
  size_t off = 0;
  bf16*  wcomb1 = (bf16*)(ws + off);  off += AL((size_t)G4 * K1 * 2);
  bf16*  whh2p  = (bf16*)(ws + off);  off += AL((size_t)G4 * KH * 2);
  bf16*  whh3p  = (bf16*)(ws + off);  off += AL((size_t)G4 * KH * 2);
  bf16*  wih2p  = (bf16*)(ws + off);  off += AL((size_t)G4 * K1 * 2);
  bf16*  wih3p  = (bf16*)(ws + off);  off += AL((size_t)G4 * K1 * 2);
  bf16*  wwinp  = (bf16*)(ws + off);  off += AL((size_t)32 * KH * 2);
  bf16*  wfcp   = (bf16*)(ws + off);  off += AL((size_t)128 * KF * 2);
  float* bias1  = (float*)(ws + off); off += AL((size_t)G4 * 4);
  bf16*  hbuf1  = (bf16*)(ws + off);  off += AL((size_t)2 * 64 * K1 * 2);
  bf16*  hb2    = (bf16*)(ws + off);  off += AL((size_t)2 * 64 * KH * 2);
  bf16*  hb3    = (bf16*)(ws + off);  off += AL((size_t)2 * 64 * KH * 2);
  int*   cidx   = (int*)(ws + off);   off += AL((size_t)64 * 64 * 4);
  int*   syncs  = (int*)(ws + off);   off += AL((size_t)4 * 4);
  bf16*  h1     = (bf16*)(ws + off);  off += AL(TB * Hdim * 2);
  bf16*  h2     = (bf16*)(ws + off);  off += AL(TB * Hdim * 2);
  bf16*  h3     = (bf16*)(ws + off);  off += AL(TB * Hdim * 2);
  bf16*  winds  = (bf16*)(ws + off);  off += AL(TB * 72 * 2);
  float* G      = (float*)(ws + off); off += AL(TB * G4 * 4);
  float* fcraw  = (float*)(ws + off); off += AL(TB * 128 * 4);

  const int mtiles = (int)(TB / 16);         // 4764

  init_kernel<<<512, 256, 0, stream>>>(stroke, hot, hbuf1, hb2, hb3, cidx, syncs);
  pack_wcomb1<<<(G4 * K1 + 255) / 256, 256, 0, stream>>>(W_ih1, W_hh1, b_ih1, b_hh1,
                                                         wcomb1, bias1);
  pad_convert<<<(G4 * KH + 255) / 256, 256, 0, stream>>>(W_hh2, whh2p, G4, Hdim, G4, KH);
  pad_convert<<<(G4 * KH + 255) / 256, 256, 0, stream>>>(W_hh3, whh3p, G4, Hdim, G4, KH);
  pad_convert<<<(G4 * K1 + 255) / 256, 256, 0, stream>>>(W_ih2, wih2p, G4, 475, G4, K1);
  pad_convert<<<(G4 * K1 + 255) / 256, 256, 0, stream>>>(W_ih3, wih3p, G4, 475, G4, K1);
  pad_convert<<<(32 * KH + 255) / 256, 256, 0, stream>>>(W_win, wwinp, 30, Hdim, 32, KH);
  pad_convert<<<(128 * KF + 255) / 256, 256, 0, stream>>>(W_fc, wfcp, 121, 1200, 128, KF);

  size_t lds1 = (size_t)2 * 64 * K1 * 2 + (size_t)32 * KH * 2 +
                (size_t)(64 * 32 + 64 * 64 + 4 * 64 * 16 + 64 * 72) * 4;
  layer1_persist<<<NWG, 128, lds1, stream>>>(stroke, wcomb1, bias1, wwinp, b_win,
                                             cidx, hbuf1, h1, winds, syncs + 0);

  pregate_gemm<<<mtiles, 256, 16 * K1 * sizeof(bf16), stream>>>(stroke, h1, winds,
                                                                wih2p, b_ih2, b_hh2, G);
  rec_persist<<<NWG, 128, (size_t)2 * 64 * KH * 2, stream>>>(G, whh2p, hb2, h2,
                                                             syncs + 1);

  pregate_gemm<<<mtiles, 256, 16 * K1 * sizeof(bf16), stream>>>(stroke, h2, winds,
                                                                wih3p, b_ih3, b_hh3, G);
  rec_persist<<<NWG, 128, (size_t)2 * 64 * KH * 2, stream>>>(G, whh3p, hb3, h3,
                                                             syncs + 2);

  fc_gemm<<<mtiles, 256, 16 * KF * sizeof(bf16), stream>>>(h1, h2, h3, wfcp, b_fc, fcraw);
  out_transform<<<(int)((TB + 255) / 256), 256, 0, stream>>>(fcraw, (float*)d_out);
}
